// AttentionBlock_26225070309665
// MI455X (gfx1250) — matrix operand order repaired and checked by reading, not compiled, not run on hardware
//
#include <hip/hip_runtime.h>

// Problem constants (match reference)
#define BB 8
#define TT 2048
#define DD 512
#define NN 512   // projection output dim (K == V == 512)

typedef __attribute__((ext_vector_type(16))) __bf16 v16bf;
typedef __attribute__((ext_vector_type(8)))  __bf16 v8bf;
typedef __attribute__((ext_vector_type(8)))  float  v8f;

static __device__ __forceinline__ v16bf cat16(v8bf lo, v8bf hi) {
    v16bf r;
#pragma unroll
    for (int i = 0; i < 8; ++i) { r[i] = lo[i]; r[i + 8] = hi[i]; }
    return r;
}

static __device__ __forceinline__ v8f wmma_bf16(v16bf a, v16bf b, v8f c) {
    // D = A(16x32 bf16) * B(32x16 bf16) + C(16x16 f32)
    return __builtin_amdgcn_wmma_f32_16x16x32_bf16(
        /*neg_a=*/false, a, /*neg_b=*/false, b,
        /*c_mod=*/(short)0, c, /*reuse_a=*/false, /*reuse_b=*/false);
}

// ---------------- fp32 -> bf16 convert of minibatch ----------------
__global__ void k_conv_x(const float* __restrict__ x, __bf16* __restrict__ xb, int n) {
    int i = blockIdx.x * 256 + threadIdx.x;
    if (i < n) xb[i] = (__bf16)x[i];
}

// ---- pack W[n][d] (row-major fp32, n=out,d=in) into WMMA B-operand layout ----
// packed index = ((ntile*(D/32) + dchunk)*32 + lane)*16 + i
// element (lane, i) holds W[n][d] with n = ntile*16 + lane%16,
//                                  d = dchunk*32 + 16*(lane/16) + i
__global__ void k_pack_w(const float* __restrict__ w, __bf16* __restrict__ wp) {
    int tid    = blockIdx.x * 256 + threadIdx.x;   // 0 .. NN*DD-1
    int i      = tid & 15;
    int lane   = (tid >> 4) & 31;
    int dchunk = (tid >> 9) & 15;                  // D/32 = 16
    int ntile  = tid >> 13;                        // NN/16 = 32
    int n = ntile * 16 + (lane & 15);
    int d = dchunk * 32 + ((lane >> 4) << 4) + i;
    wp[tid] = (__bf16)w[n * DD + d];
}

// ---------------- projection GEMM: Y = X * W^T + b ----------------
// One wave -> one 32x32 output tile (2x2 WMMA register blocking).
// transposeOut=1 stores [b][n][t] (for V).
__global__ void k_proj(const __bf16* __restrict__ xb, const __bf16* __restrict__ wp,
                       const float* __restrict__ bias, __bf16* __restrict__ out,
                       int transposeOut) {
    int lane = threadIdx.x & 31;
    int wave = threadIdx.x >> 5;
    int g     = blockIdx.x * 4 + wave;   // 512 * 16 tiles
    int nTile = g & 15;                  // NN/32
    int tRow  = g >> 4;                  // (B*T)/32
    int t0 = tRow * 32;
    int n0 = nTile * 32;
    int g2 = lane >> 4, ln = lane & 15;

    const __bf16* arow0 = xb + (size_t)(t0 + ln) * DD;
    const __bf16* arow1 = xb + (size_t)(t0 + 16 + ln) * DD;
    const __bf16* bb0   = wp + (size_t)(n0 >> 4) * 8192;        // 16 chunks * 32 lanes * 16
    const __bf16* bb1   = bb0 + 8192;

    v8f c00 = {}, c01 = {}, c10 = {}, c11 = {};
#pragma unroll 2
    for (int dc = 0; dc < 16; ++dc) {
        int d0 = dc * 32;
        v16bf a0 = cat16(*(const v8bf*)(arow0 + d0 + 16 * g2),
                         *(const v8bf*)(arow0 + d0 + 8 + 16 * g2));
        v16bf a1 = cat16(*(const v8bf*)(arow1 + d0 + 16 * g2),
                         *(const v8bf*)(arow1 + d0 + 8 + 16 * g2));
        v16bf b0 = *(const v16bf*)(bb0 + (size_t)dc * 512 + lane * 16);
        v16bf b1 = *(const v16bf*)(bb1 + (size_t)dc * 512 + lane * 16);
        c00 = wmma_bf16(a0, b0, c00);
        c01 = wmma_bf16(a0, b1, c01);
        c10 = wmma_bf16(a1, b0, c10);
        c11 = wmma_bf16(a1, b1, c11);
    }

    float bn0 = bias[n0 + ln];
    float bn1 = bias[n0 + 16 + ln];
    if (transposeOut) {
        int b_  = t0 / TT;
        int tl0 = t0 - b_ * TT;
        __bf16* o0 = out + (size_t)b_ * NN * TT + (size_t)(n0 + ln) * TT;
        __bf16* o1 = out + (size_t)b_ * NN * TT + (size_t)(n0 + 16 + ln) * TT;
#pragma unroll
        for (int r = 0; r < 8; ++r) {
            int rr = r + 8 * g2;
            o0[tl0 + rr]      = (__bf16)(c00[r] + bn0);
            o1[tl0 + rr]      = (__bf16)(c01[r] + bn1);
            o0[tl0 + 16 + rr] = (__bf16)(c10[r] + bn0);
            o1[tl0 + 16 + rr] = (__bf16)(c11[r] + bn1);
        }
    } else {
#pragma unroll
        for (int r = 0; r < 8; ++r) {
            int rr = r + 8 * g2;
            out[(size_t)(t0 + rr) * NN + n0 + ln]           = (__bf16)(c00[r] + bn0);
            out[(size_t)(t0 + rr) * NN + n0 + 16 + ln]      = (__bf16)(c01[r] + bn1);
            out[(size_t)(t0 + 16 + rr) * NN + n0 + ln]      = (__bf16)(c10[r] + bn0);
            out[(size_t)(t0 + 16 + rr) * NN + n0 + 16 + ln] = (__bf16)(c11[r] + bn1);
        }
    }
}

// ---------------- logits = Q * K^T (per batch), fp32 out; masked tiles skipped ----------------
// One wave -> 32x32 logits tile.
__global__ void k_logits(const __bf16* __restrict__ qb, const __bf16* __restrict__ kb,
                         float* __restrict__ logits) {
    int lane = threadIdx.x & 31;
    int wave = threadIdx.x >> 5;
    int g  = blockIdx.x * 4 + wave;      // 8*64*64 tiles
    int si = g & 63;
    int rem = g >> 6;
    int ti = rem & 63;
    int b  = rem >> 6;
    if (si > ti) return;                 // fully masked (s0 > t0+31): never read downstream
    int t0 = ti * 32, s0 = si * 32;
    int g2 = lane >> 4, ln = lane & 15;

    const __bf16* qrow0 = qb + ((size_t)b * TT + t0 + ln) * NN;
    const __bf16* qrow1 = qb + ((size_t)b * TT + t0 + 16 + ln) * NN;
    const __bf16* krow0 = kb + ((size_t)b * TT + s0 + ln) * NN;       // row-major K == B layout
    const __bf16* krow1 = kb + ((size_t)b * TT + s0 + 16 + ln) * NN;

    v8f c00 = {}, c01 = {}, c10 = {}, c11 = {};
#pragma unroll 2
    for (int kc = 0; kc < 16; ++kc) {
        int k0 = kc * 32;
        v16bf a0 = cat16(*(const v8bf*)(qrow0 + k0 + 16 * g2),
                         *(const v8bf*)(qrow0 + k0 + 8 + 16 * g2));
        v16bf a1 = cat16(*(const v8bf*)(qrow1 + k0 + 16 * g2),
                         *(const v8bf*)(qrow1 + k0 + 8 + 16 * g2));
        v16bf b0 = *(const v16bf*)(krow0 + k0 + 16 * g2);
        v16bf b1 = *(const v16bf*)(krow1 + k0 + 16 * g2);
        c00 = wmma_bf16(a0, b0, c00);
        c01 = wmma_bf16(a0, b1, c01);
        c10 = wmma_bf16(a1, b0, c10);
        c11 = wmma_bf16(a1, b1, c11);
    }
    float* lb = logits + (size_t)b * TT * TT;
#pragma unroll
    for (int r = 0; r < 8; ++r) {
        int rr = r + 8 * g2;
        lb[(size_t)(t0 + rr) * TT + s0 + ln]           = c00[r];
        lb[(size_t)(t0 + rr) * TT + s0 + 16 + ln]      = c01[r];
        lb[(size_t)(t0 + 16 + rr) * TT + s0 + ln]      = c10[r];
        lb[(size_t)(t0 + 16 + rr) * TT + s0 + 16 + ln] = c11[r];
    }
}

// ------------- column-wise online softmax stats: max & sum over t in [s, T) -------------
__global__ void k_stats(const float* __restrict__ logits, float* __restrict__ cmax,
                        float* __restrict__ csum) {
    int tid = blockIdx.x * 256 + threadIdx.x;   // B*T threads, one (b,s) column each
    int b = tid >> 11, s = tid & (TT - 1);
    const float* lb = logits + (size_t)b * TT * TT;
    float m = -3.4e38f, sum = 0.f;
    for (int t = s; t < TT; ++t) {
        float v = lb[(size_t)t * TT + s];
        if (v > m) { sum = sum * __expf(m - v) + 1.f; m = v; }
        else       { sum += __expf(v - m); }
    }
    cmax[tid] = m;
    csum[tid] = sum;
}

// ------------- probs (bf16) = exp(l - colmax)/colsum / sqrt(512); 0 where masked -------------
__global__ void k_probs(const float* __restrict__ logits, const float* __restrict__ cmax,
                        const float* __restrict__ csum, __bf16* __restrict__ probs) {
    long long tid = (long long)blockIdx.x * 256 + threadIdx.x;  // enumerates [b][t][s]
    int b   = (int)(tid >> 22);
    int rem = (int)(tid & ((long long)TT * TT - 1));
    int t = rem >> 11, s = rem & (TT - 1);
    float p = 0.f;
    if (s <= t) {
        float v = logits[tid];
        int cs = (b << 11) + s;
        p = __expf(v - cmax[cs]) / csum[cs] * 0.04419417382415922f;  // 1/sqrt(512)
    }
    probs[tid] = (__bf16)p;
}

// ------------- read = probs * V (V transposed [b][v][t]); out = minibatch + read -------------
// One wave -> 32x32 output tile.
__global__ void k_read(const __bf16* __restrict__ probs, const __bf16* __restrict__ vt,
                       const float* __restrict__ x, float* __restrict__ out) {
    int lane = threadIdx.x & 31;
    int wave = threadIdx.x >> 5;
    int g  = blockIdx.x * 4 + wave;     // 8*64*16 tiles
    int vi = g & 15;
    int rem = g >> 4;
    int ti = rem & 63;
    int b  = rem >> 6;
    int t0 = ti * 32, v0 = vi * 32;
    int g2 = lane >> 4, ln = lane & 15;

    const __bf16* prow0 = probs + ((size_t)b * TT + t0 + ln) * TT;
    const __bf16* prow1 = probs + ((size_t)b * TT + t0 + 16 + ln) * TT;
    const __bf16* vrow0 = vt + (size_t)b * NN * TT + (size_t)(v0 + ln) * TT;
    const __bf16* vrow1 = vt + (size_t)b * NN * TT + (size_t)(v0 + 16 + ln) * TT;

    int nchunk = (t0 >> 5) + 1;         // causal: only s <= t0+31 can be nonzero
    v8f c00 = {}, c01 = {}, c10 = {}, c11 = {};
    for (int sc = 0; sc < nchunk; ++sc) {
        int s0 = sc * 32;
        v16bf a0 = cat16(*(const v8bf*)(prow0 + s0 + 16 * g2),
                         *(const v8bf*)(prow0 + s0 + 8 + 16 * g2));
        v16bf a1 = cat16(*(const v8bf*)(prow1 + s0 + 16 * g2),
                         *(const v8bf*)(prow1 + s0 + 8 + 16 * g2));
        v16bf b0 = *(const v16bf*)(vrow0 + s0 + 16 * g2);
        v16bf b1 = *(const v16bf*)(vrow1 + s0 + 16 * g2);
        c00 = wmma_bf16(a0, b0, c00);
        c01 = wmma_bf16(a0, b1, c01);
        c10 = wmma_bf16(a1, b0, c10);
        c11 = wmma_bf16(a1, b1, c11);
    }
#pragma unroll
    for (int r = 0; r < 8; ++r) {
        int rr = r + 8 * g2;
        size_t i00 = ((size_t)b * TT + t0 + rr) * DD + v0 + ln;
        size_t i01 = ((size_t)b * TT + t0 + rr) * DD + v0 + 16 + ln;
        size_t i10 = ((size_t)b * TT + t0 + 16 + rr) * DD + v0 + ln;
        size_t i11 = ((size_t)b * TT + t0 + 16 + rr) * DD + v0 + 16 + ln;
        out[i00] = x[i00] + c00[r];
        out[i01] = x[i01] + c01[r];
        out[i10] = x[i10] + c10[r];
        out[i11] = x[i11] + c11[r];
    }
}

extern "C" void kernel_launch(void* const* d_in, const int* in_sizes, int n_in,
                              void* d_out, int out_size, void* d_ws, size_t ws_size,
                              hipStream_t stream) {
    const float* mb = (const float*)d_in[0];
    const float* Wk = (const float*)d_in[1];
    const float* bk = (const float*)d_in[2];
    const float* Wq = (const float*)d_in[3];
    const float* bq = (const float*)d_in[4];
    const float* Wv = (const float*)d_in[5];
    const float* bv = (const float*)d_in[6];
    float* out = (float*)d_out;

    char* ws = (char*)d_ws;
    size_t off = 0;
    auto alloc = [&](size_t bytes) {
        size_t o = off;
        off += (bytes + 255) & ~(size_t)255;
        return o;
    };
    __bf16* Xb     = (__bf16*)(ws + alloc((size_t)BB * TT * DD * 2));
    __bf16* Wkp    = (__bf16*)(ws + alloc((size_t)NN * DD * 2));
    __bf16* Wqp    = (__bf16*)(ws + alloc((size_t)NN * DD * 2));
    __bf16* Wvp    = (__bf16*)(ws + alloc((size_t)NN * DD * 2));
    __bf16* Qb     = (__bf16*)(ws + alloc((size_t)BB * TT * NN * 2));
    __bf16* Kb     = (__bf16*)(ws + alloc((size_t)BB * TT * NN * 2));
    __bf16* Vt     = (__bf16*)(ws + alloc((size_t)BB * NN * TT * 2));  // transposed V
    float*  logits = (float*)(ws + alloc((size_t)BB * TT * TT * 4));
    float*  cmax   = (float*)(ws + alloc((size_t)BB * TT * 4));
    float*  csum   = (float*)(ws + alloc((size_t)BB * TT * 4));
    __bf16* probs  = (__bf16*)(ws + alloc((size_t)BB * TT * TT * 2));

    (void)in_sizes; (void)n_in; (void)out_size; (void)ws_size;

    // 1) convert inputs to bf16 / packed-B layouts
    k_conv_x<<<(BB * TT * DD) / 256, 256, 0, stream>>>(mb, Xb, BB * TT * DD);
    k_pack_w<<<(NN * DD) / 256, 256, 0, stream>>>(Wk, Wkp);
    k_pack_w<<<(NN * DD) / 256, 256, 0, stream>>>(Wq, Wqp);
    k_pack_w<<<(NN * DD) / 256, 256, 0, stream>>>(Wv, Wvp);

    // 2) projections (WMMA, 32x32/wave): Q, K row-major bf16; V transposed bf16
    k_proj<<<(512 * 16) / 4, 128, 0, stream>>>(Xb, Wqp, bq, Qb, 0);
    k_proj<<<(512 * 16) / 4, 128, 0, stream>>>(Xb, Wkp, bk, Kb, 0);
    k_proj<<<(512 * 16) / 4, 128, 0, stream>>>(Xb, Wvp, bv, Vt, 1);

    // 3) causal logits (WMMA, 32x32/wave), fp32
    k_logits<<<(BB * 64 * 64) / 4, 128, 0, stream>>>(Qb, Kb, logits);

    // 4) column-wise softmax stats + probs (bf16)
    k_stats<<<(BB * TT) / 256, 256, 0, stream>>>(logits, cmax, csum);
    k_probs<<<(BB * TT * (TT / 256)), 256, 0, stream>>>(logits, cmax, csum, probs);

    // 5) read = probs * V (WMMA, 32x32/wave) + residual, fp32 out
    k_read<<<(BB * 64 * 16) / 4, 128, 0, stream>>>(probs, Vt, mb, out);
}
